// TransformerBlockM2_86346022519156
// MI455X (gfx1250) — matrix operand order repaired and checked by reading, not compiled, not run on hardware
//
#include <hip/hip_runtime.h>

// Problem constants (match reference)
#define B_    512
#define S_    193
#define D_    64
#define H_    2
#define DK_   32
#define DFF_  256
#define SP_   224                 // padded sequence (7 * 32)
#define BS_   (B_ * S_)           // 98816 (divisible by 16)
#define NTILES_ (BS_ / 16)        // 6176

typedef __attribute__((ext_vector_type(16))) _Float16 v16h;
typedef __attribute__((ext_vector_type(8)))  _Float16 v8h;
typedef __attribute__((ext_vector_type(8)))  float    v8f;
typedef __attribute__((ext_vector_type(4)))  float    v4f;

// ---------------------------------------------------------------------------
// WMMA operand builders (wave32; contiguous K order used for A and B alike).
// A (16x32 f16): lane holds row M=lane&15; its 16 K-values are contiguous:
// K = koff..koff+15, koff = 16*(lane>=16).
// B (32x16 f16): lane holds column N=lane&15; its 16 K-values are contiguous:
// K = kb..kb+15, kb = 16*(lane>=16)  -> one 32-byte load from K-major storage.
// ---------------------------------------------------------------------------
__device__ __forceinline__ v16h a_from_f32(const float* __restrict__ p, int lane) {
  const int koff = (lane & 16) ? 16 : 0;
  v4f f0 = *(const v4f*)(p + koff);
  v4f f1 = *(const v4f*)(p + koff + 4);
  v4f f2 = *(const v4f*)(p + koff + 8);
  v4f f3 = *(const v4f*)(p + koff + 12);
  v16h a;
  #pragma unroll
  for (int i = 0; i < 4; ++i) {
    a[i]      = (_Float16)f0[i];
    a[4 + i]  = (_Float16)f1[i];
    a[8 + i]  = (_Float16)f2[i];
    a[12 + i] = (_Float16)f3[i];
  }
  return a;
}

__device__ __forceinline__ v16h a_from_h16(const _Float16* __restrict__ p, int lane) {
  const int koff = (lane & 16) ? 16 : 0;
  v8h lo = *(const v8h*)(p + koff);
  v8h hi = *(const v8h*)(p + koff + 8);
  v16h a;
  #pragma unroll
  for (int i = 0; i < 8; ++i) { a[i] = lo[i]; a[8 + i] = hi[i]; }
  return a;
}

// ---------------------------------------------------------------------------
// Weight prep: transpose f32 W[K][N] -> f16 WT[N][K] (K-major columns so that
// B tiles are contiguous 32-byte lane loads).
// ---------------------------------------------------------------------------
__global__ void wprep_kernel(const float* __restrict__ src, _Float16* __restrict__ dst,
                             int K, int N) {
  int idx = blockIdx.x * 256 + threadIdx.x;
  if (idx < K * N) {
    int n = idx / K, k = idx - n * K;
    dst[idx] = (_Float16)src[(size_t)k * N + n];
  }
}

// ---------------------------------------------------------------------------
// Kernel 1: gated QKV projections.
// grid = NTILES_, block = 128 (4 waves); wave w owns output N-tile w.
// q,k stored f16 [B,H,SP,DK]; v stored transposed f16 [B,H,DK,SP].
// ---------------------------------------------------------------------------
__global__ void qkv_proj_kernel(
    const float* __restrict__ state, const float* __restrict__ C1, const float* __restrict__ C2,
    const _Float16* __restrict__ WqT, const _Float16* __restrict__ WkT, const _Float16* __restrict__ WvT,
    const _Float16* __restrict__ WCqT, const _Float16* __restrict__ WCkT, const _Float16* __restrict__ WCvT,
    const float* __restrict__ bq,  const float* __restrict__ bk,  const float* __restrict__ bv,
    const float* __restrict__ bCq, const float* __restrict__ bCk, const float* __restrict__ bCv,
    const float* __restrict__ gQE, const float* __restrict__ bQE,
    const float* __restrict__ gKE, const float* __restrict__ bKE,
    const float* __restrict__ gVE, const float* __restrict__ bVE,
    const float* __restrict__ gQH, const float* __restrict__ bQH,
    const float* __restrict__ gKH, const float* __restrict__ bKH,
    const float* __restrict__ gVH, const float* __restrict__ bVH,
    _Float16* __restrict__ qh, _Float16* __restrict__ kh, _Float16* __restrict__ vT)
{
  __shared__ float Ebuf[3][16][64];
  __shared__ float Hbuf[3][16][64];

  const int lane = threadIdx.x & 31;
  const int wave = threadIdx.x >> 5;        // N-tile id 0..3
  const int row0 = blockIdx.x * 16;
  const int arow = lane & 15;
  const int bcol = lane & 15;
  const int kb   = (lane & 16) ? 16 : 0;
  const int mb   = (lane & 16) ? 8  : 0;

  const float* bE[3] = {bq, bk, bv};
  const float* bH[3] = {bCq, bCk, bCv};

  const int ncol = wave * 16 + bcol;        // this lane's B column

  // hoisted per-lane operand base pointers
  const float* srow  = state + (size_t)(row0 + arow) * D_;
  const float* c1row = C1 + (size_t)(row0 + arow) * DFF_;
  const float* c2row = C2 + (size_t)(row0 + arow) * DFF_;
  const _Float16* wE[3] = {WqT  + (size_t)ncol * D_ + kb,
                           WkT  + (size_t)ncol * D_ + kb,
                           WvT  + (size_t)ncol * D_ + kb};
  const _Float16* wH[3] = {WCqT + (size_t)ncol * (2 * DFF_) + kb,
                           WCkT + (size_t)ncol * (2 * DFF_) + kb,
                           WCvT + (size_t)ncol * (2 * DFF_) + kb};

  // state branch: K = 64 (2 chunks)
  v8f accE[3] = {};
  #pragma unroll
  for (int kc = 0; kc < 2; ++kc) {
    v16h aS = a_from_f32(srow + kc * 32, lane);
    #pragma unroll
    for (int p = 0; p < 3; ++p) {
      v16h bW = *(const v16h*)(wE[p] + kc * 32);
      accE[p] = __builtin_amdgcn_wmma_f32_16x16x32_f16(false, aS, false, bW,
                                                       (short)0, accE[p], false, false);
    }
  }

  // context branch: K = 512; chunks 0..7 from C1, 8..15 from C2 (no selects)
  v8f accH[3] = {};
  for (int kc = 0; kc < 8; ++kc) {
    v16h aC = a_from_f32(c1row + kc * 32, lane);
    #pragma unroll
    for (int p = 0; p < 3; ++p) {
      v16h bW = *(const v16h*)(wH[p] + kc * 32);
      accH[p] = __builtin_amdgcn_wmma_f32_16x16x32_f16(false, aC, false, bW,
                                                       (short)0, accH[p], false, false);
    }
  }
  for (int kc = 0; kc < 8; ++kc) {
    v16h aC = a_from_f32(c2row + kc * 32, lane);
    #pragma unroll
    for (int p = 0; p < 3; ++p) {
      v16h bW = *(const v16h*)(wH[p] + DFF_ + kc * 32);
      accH[p] = __builtin_amdgcn_wmma_f32_16x16x32_f16(false, aC, false, bW,
                                                       (short)0, accH[p], false, false);
    }
  }

  // spill (+bias) to LDS for per-row LayerNorm
  #pragma unroll
  for (int p = 0; p < 3; ++p)
    #pragma unroll
    for (int r = 0; r < 8; ++r) {
      Ebuf[p][mb + r][ncol] = accE[p][r] + bE[p][ncol];
      Hbuf[p][mb + r][ncol] = accH[p][r] + bH[p][ncol];
    }
  __syncthreads();

  const float* gEl[3] = {gQE, gKE, gVE};
  const float* bEl[3] = {bQE, bKE, bVE};
  const float* gHl[3] = {gQH, gKH, gVH};
  const float* bHl[3] = {bQH, bKH, bVH};

  const int t = threadIdx.x;
  if (t < 48) {
    const int p = t >> 4, row = t & 15;
    float se = 0.f, se2 = 0.f, sh = 0.f, sh2 = 0.f;
    for (int c = 0; c < 64; ++c) {
      float e = Ebuf[p][row][c];  se += e;  se2 += e * e;
      float x = Hbuf[p][row][c];  sh += x;  sh2 += x * x;
    }
    const float me = se / 64.f, mhm = sh / 64.f;
    const float rse = rsqrtf(se2 / 64.f - me * me + 1e-5f);
    const float rsh = rsqrtf(sh2 / 64.f - mhm * mhm + 1e-5f);
    const int gr = row0 + row;
    const int bb = gr / S_, ss = gr % S_;
    for (int c = 0; c < 64; ++c) {
      float e = (Ebuf[p][row][c] - me)  * rse * gEl[p][c] + bEl[p][c];
      float x = (Hbuf[p][row][c] - mhm) * rsh * gHl[p][c] + bHl[p][c];
      const int hd = c >> 5, dk = c & 31;
      const _Float16 val = (_Float16)(e * x);
      if (p == 0)       qh[(size_t)((bb * H_ + hd) * SP_ + ss) * DK_ + dk] = val;
      else if (p == 1)  kh[(size_t)((bb * H_ + hd) * SP_ + ss) * DK_ + dk] = val;
      else              vT[(size_t)((bb * H_ + hd) * DK_ + dk) * SP_ + ss] = val;
    }
  }
}

// ---------------------------------------------------------------------------
// Kernel 2: attention. grid = B*H*13 (one wave per (b,h,q-tile)), block = 32.
// ---------------------------------------------------------------------------
__global__ void attn_kernel(const _Float16* __restrict__ qh,
                            const _Float16* __restrict__ kh,
                            const _Float16* __restrict__ vT,
                            float* __restrict__ o)
{
  __shared__ float    Sc[16 * SP_];
  __shared__ _Float16 Ph[16 * SP_];
  __shared__ float    red[2][16];
  __shared__ float    invs[16];

  const int lane = threadIdx.x & 31;
  const int bh = blockIdx.x / 13;
  const int qt = blockIdx.x % 13;
  const int bb = bh / H_, hh = bh % H_;
  const size_t base = (size_t)bh * SP_ * DK_;

  const int arow = lane & 15;
  const int bcol = lane & 15;
  const int kb   = (lane & 16) ? 16 : 0;
  const int mb   = (lane & 16) ? 8  : 0;

  // Q tile (16x32): two contiguous 8-half runs per lane
  v16h aQ = a_from_h16(qh + base + (size_t)(qt * 16 + arow) * DK_, lane);

  const v8f zero = {};
  #pragma unroll
  for (int jt = 0; jt < SP_ / 16; ++jt) {
    v16h bK = *(const v16h*)(kh + base + (size_t)(jt * 16 + bcol) * DK_ + kb);
    v8f sc = __builtin_amdgcn_wmma_f32_16x16x32_f16(false, aQ, false, bK,
                                                    (short)0, zero, false, false);
    #pragma unroll
    for (int r = 0; r < 8; ++r)
      Sc[(mb + r) * SP_ + jt * 16 + bcol] = sc[r] * 0.17677669529663687f; // 1/sqrt(32)
  }
  __syncthreads();

  // row softmax: two lanes per row, each covering half the (padded) keys
  {
    const int row  = lane & 15;
    const int half = lane >> 4;
    const int j0   = half * (SP_ / 2);
    float mx = -1e30f;
    for (int j = j0; j < j0 + SP_ / 2; ++j)
      if (j < S_) mx = fmaxf(mx, Sc[row * SP_ + j]);
    red[half][row] = mx;
    __syncthreads();
    mx = fmaxf(red[0][row], red[1][row]);
    __syncthreads();
    float sum = 0.f;
    for (int j = j0; j < j0 + SP_ / 2; ++j) {
      float e = 0.f;
      if (j < S_) e = __expf(Sc[row * SP_ + j] - mx);
      sum += e;
      Ph[row * SP_ + j] = (_Float16)e;
    }
    red[half][row] = sum;
    __syncthreads();
    if (lane < 16) invs[row] = 1.f / (red[0][row] + red[1][row]);
  }
  __syncthreads();

  // O = P @ V : K = 224 (7 chunks), two N-tiles (DK=32). V is K-major (vT).
  v8f acc0 = {}, acc1 = {};
  for (int kc = 0; kc < SP_ / 32; ++kc) {
    v16h aP = a_from_h16(&Ph[arow * SP_ + kc * 32], lane);
    v16h bV0 = *(const v16h*)(vT + ((size_t)bh * DK_ + bcol)      * SP_ + kc * 32 + kb);
    v16h bV1 = *(const v16h*)(vT + ((size_t)bh * DK_ + 16 + bcol) * SP_ + kc * 32 + kb);
    acc0 = __builtin_amdgcn_wmma_f32_16x16x32_f16(false, aP, false, bV0, (short)0, acc0, false, false);
    acc1 = __builtin_amdgcn_wmma_f32_16x16x32_f16(false, aP, false, bV1, (short)0, acc1, false, false);
  }
  #pragma unroll
  for (int r = 0; r < 8; ++r) {
    const int M = mb + r;
    const int qrow = qt * 16 + M;
    if (qrow < S_) {
      const float inv = invs[M];
      const size_t oidx = (size_t)(bb * S_ + qrow) * D_ + hh * DK_;
      o[oidx + bcol]      = acc0[r] * inv;
      o[oidx + 16 + bcol] = acc1[r] * inv;
    }
  }
}

// ---------------------------------------------------------------------------
// Kernel 3: residual + LN + FFN + outputs. grid = NTILES_, block = 128.
// ---------------------------------------------------------------------------
__global__ void ffn_kernel(const float* __restrict__ state,
                           const float* __restrict__ o,
                           const _Float16* __restrict__ W1T,  // [256][64]
                           const float* __restrict__ b1,
                           const _Float16* __restrict__ W2T,  // [64][256]
                           const float* __restrict__ b2,
                           const float* __restrict__ g1, const float* __restrict__ be1,
                           float* __restrict__ outZ5, float* __restrict__ outDZ,
                           float* __restrict__ outSt)
{
  __shared__ float    Z1f[16][64];
  __shared__ float    mrs[16][2];
  __shared__ _Float16 Z2h[16][64];
  __shared__ _Float16 Z3h[16][256];

  const int tid  = threadIdx.x;
  const int lane = tid & 31;
  const int wave = tid >> 5;
  const int row0 = blockIdx.x * 16;
  const int arow = lane & 15;
  const int bcol = lane & 15;
  const int kb   = (lane & 16) ? 16 : 0;
  const int mb   = (lane & 16) ? 8  : 0;

  // Z1 = state + o (coalesced into LDS)
  #pragma unroll
  for (int it = 0; it < 8; ++it) {
    const int idx = it * 128 + tid;            // 0..1023
    const int row = idx >> 6, c = idx & 63;
    const size_t gidx = (size_t)(row0 + row) * D_ + c;
    Z1f[row][c] = state[gidx] + o[gidx];
  }
  __syncthreads();

  if (tid < 16) {
    float s = 0.f, s2 = 0.f;
    for (int c = 0; c < 64; ++c) { const float v = Z1f[tid][c]; s += v; s2 += v * v; }
    const float m = s / 64.f;
    mrs[tid][0] = m;
    mrs[tid][1] = rsqrtf(s2 / 64.f - m * m + 1e-5f);
  }
  __syncthreads();

  #pragma unroll
  for (int it = 0; it < 8; ++it) {
    const int idx = it * 128 + tid;
    const int row = idx >> 6, c = idx & 63;
    Z2h[row][c] = (_Float16)((Z1f[row][c] - mrs[row][0]) * mrs[row][1] * g1[c] + be1[c]);
  }
  __syncthreads();

  // Z3 = gelu_tanh(Z2 @ W1 + b1): wave covers 4 of 16 N-tiles
  #pragma unroll
  for (int tt = 0; tt < 4; ++tt) {
    const int nt = wave * 4 + tt;
    v8f acc = {};
    #pragma unroll
    for (int kc = 0; kc < 2; ++kc) {
      v16h aA = a_from_h16(&Z2h[arow][kc * 32], lane);
      v16h bW = *(const v16h*)(W1T + (size_t)(nt * 16 + bcol) * D_ + kc * 32 + kb);
      acc = __builtin_amdgcn_wmma_f32_16x16x32_f16(false, aA, false, bW, (short)0, acc, false, false);
    }
    const float bias = b1[nt * 16 + bcol];
    #pragma unroll
    for (int r = 0; r < 8; ++r) {
      const float x = acc[r] + bias;
      const float g = 0.5f * x * (1.f + tanhf(0.7978845608028654f * (x + 0.044715f * x * x * x)));
      Z3h[mb + r][nt * 16 + bcol] = (_Float16)g;
    }
  }
  __syncthreads();

  // Z4 = Z3 @ W2 + b2: each wave one of 4 output N-tiles
  v8f acc = {};
  #pragma unroll
  for (int kc = 0; kc < 8; ++kc) {
    v16h aA = a_from_h16(&Z3h[arow][kc * 32], lane);
    v16h bW = *(const v16h*)(W2T + (size_t)(wave * 16 + bcol) * (4 * D_) + kc * 32 + kb);
    acc = __builtin_amdgcn_wmma_f32_16x16x32_f16(false, aA, false, bW, (short)0, acc, false, false);
  }
  const float bias2 = b2[wave * 16 + bcol];
  #pragma unroll
  for (int r = 0; r < 8; ++r) {
    const size_t idx = (size_t)(row0 + mb + r) * D_ + wave * 16 + bcol;
    const float ov = o[idx];
    const float st = state[idx];
    const float dz = ov + acc[r] + bias2;
    outZ5[idx] = st + dz;
    outDZ[idx] = dz;
    outSt[idx] = st;
  }
}

// ---------------------------------------------------------------------------
extern "C" void kernel_launch(void* const* d_in, const int* in_sizes, int n_in,
                              void* d_out, int out_size, void* d_ws, size_t ws_size,
                              hipStream_t stream) {
  const float* state = (const float*)d_in[0];
  const float* C1    = (const float*)d_in[1];
  const float* C2    = (const float*)d_in[2];
  const float* Wq  = (const float*)d_in[3];  const float* bq  = (const float*)d_in[4];
  const float* Wk  = (const float*)d_in[5];  const float* bk  = (const float*)d_in[6];
  const float* Wv  = (const float*)d_in[7];  const float* bv  = (const float*)d_in[8];
  const float* WCq = (const float*)d_in[9];  const float* bCq = (const float*)d_in[10];
  const float* WCk = (const float*)d_in[11]; const float* bCk = (const float*)d_in[12];
  const float* WCv = (const float*)d_in[13]; const float* bCv = (const float*)d_in[14];
  const float* gQE = (const float*)d_in[15]; const float* bQE = (const float*)d_in[16];
  const float* gKE = (const float*)d_in[17]; const float* bKE = (const float*)d_in[18];
  const float* gVE = (const float*)d_in[19]; const float* bVE = (const float*)d_in[20];
  const float* gQH = (const float*)d_in[21]; const float* bQH = (const float*)d_in[22];
  const float* gKH = (const float*)d_in[23]; const float* bKH = (const float*)d_in[24];
  const float* gVH = (const float*)d_in[25]; const float* bVH = (const float*)d_in[26];
  const float* W1  = (const float*)d_in[27]; const float* b1  = (const float*)d_in[28];
  const float* W2  = (const float*)d_in[29]; const float* b2  = (const float*)d_in[30];
  const float* g1  = (const float*)d_in[31]; const float* be1 = (const float*)d_in[32];

  const size_t qkvElems = (size_t)B_ * H_ * SP_ * DK_;
  _Float16* qh = (_Float16*)d_ws;
  _Float16* kh = qh + qkvElems;
  _Float16* vT = kh + qkvElems;
  float*    o  = (float*)(vT + qkvElems);

  _Float16* wts  = (_Float16*)(o + (size_t)BS_ * D_);
  _Float16* WqT  = wts;                       // 64*64
  _Float16* WkT  = WqT + 64 * 64;
  _Float16* WvT  = WkT + 64 * 64;
  _Float16* WCqT = WvT + 64 * 64;             // 64*512
  _Float16* WCkT = WCqT + (size_t)64 * 512;
  _Float16* WCvT = WCkT + (size_t)64 * 512;
  _Float16* W1T  = WCvT + (size_t)64 * 512;   // 256*64
  _Float16* W2T  = W1T + (size_t)256 * 64;    // 64*256

  float* out   = (float*)d_out;
  float* outZ5 = out;
  float* outDZ = out + (size_t)BS_ * D_;
  float* outSt = out + 2 * (size_t)BS_ * D_;

  // zero q/k/vT (including sequence padding) — graph-capture-safe
  hipMemsetAsync(d_ws, 0, qkvElems * 3 * sizeof(_Float16), stream);

  // weight transpose + f16 conversion (tiny)
  auto tp = [&](const float* s, _Float16* dvec, int K, int N) {
    const int n = K * N;
    wprep_kernel<<<(n + 255) / 256, 256, 0, stream>>>(s, dvec, K, N);
  };
  tp(Wq,  WqT,  64, 64);   tp(Wk,  WkT,  64, 64);   tp(Wv,  WvT,  64, 64);
  tp(WCq, WCqT, 512, 64);  tp(WCk, WCkT, 512, 64);  tp(WCv, WCvT, 512, 64);
  tp(W1,  W1T,  64, 256);  tp(W2,  W2T,  256, 64);

  qkv_proj_kernel<<<NTILES_, 128, 0, stream>>>(
      state, C1, C2, WqT, WkT, WvT, WCqT, WCkT, WCvT,
      bq, bk, bv, bCq, bCk, bCv,
      gQE, bQE, gKE, bKE, gVE, bVE, gQH, bQH, gKH, bKH, gVH, bVH,
      qh, kh, vT);

  attn_kernel<<<B_ * H_ * 13, 32, 0, stream>>>(qh, kh, vT, o);

  ffn_kernel<<<NTILES_, 128, 0, stream>>>(
      state, o, W1T, b1, W2T, b2, g1, be1, outZ5, outDZ, outSt);
}
